// Equilibrium_Propegation_Network_39161511805203
// MI455X (gfx1250) — matrix operand order repaired and checked by reading, not compiled, not run on hardware
//
#include <hip/hip_runtime.h>
#include <hip/hip_bf16.h>
#include <math.h>
#include <stdint.h>

typedef __attribute__((ext_vector_type(16))) _Float16 v16h;
typedef __attribute__((ext_vector_type(8)))  _Float16 v8h;
typedef __attribute__((ext_vector_type(8)))  float    v8f;

#define B_SZ   8192
#define I_SZ   1024
#define H_SZ   128
#define O_SZ   1000
#define O_PAD  1024
#define EPS_LR 0.5f

// ---------------------------------------------------------------------------
// rho(s)=clip(s,0,1). jnp.clip = minimum(1, maximum(0, s)); lax.max/min split
// the gradient 0.5/0.5 at exact ties, 1 strictly inside, 0 outside.
__device__ __forceinline__ float rho_prime(float s) {
    return (s > 0.0f && s < 1.0f) ? 1.0f
         : ((s == 0.0f || s == 1.0f) ? 0.5f : 0.0f);
}

// CDNA5 async global->LDS copy (ASYNCcnt-tracked, bypasses VGPRs).
__device__ __forceinline__ void async_b128(unsigned ldsOff, unsigned gOff, uint64_t base) {
    asm volatile("global_load_async_to_lds_b128 %0, %1, %2"
                 :: "v"(ldsOff), "v"(gOff), "s"(base) : "memory");
}
__device__ __forceinline__ void wait_async0() {
    asm volatile("s_wait_asynccnt 0x0" ::: "memory");
}

// 16x32 f16 A-fragment from an f32 row (used once, for x@W1).
// Lanes 0-15 hold row M=l15 with K = klo..klo+15.
__device__ __forceinline__ v16h load_a_from_f32(const float* __restrict__ rowPtr, int klo) {
    const float4 f0 = *(const float4*)(rowPtr + klo);
    const float4 f1 = *(const float4*)(rowPtr + klo + 4);
    const float4 f2 = *(const float4*)(rowPtr + klo + 8);
    const float4 f3 = *(const float4*)(rowPtr + klo + 12);
    v16h a;
    a[0]  = (_Float16)f0.x; a[1]  = (_Float16)f0.y; a[2]  = (_Float16)f0.z; a[3]  = (_Float16)f0.w;
    a[4]  = (_Float16)f1.x; a[5]  = (_Float16)f1.y; a[6]  = (_Float16)f1.z; a[7]  = (_Float16)f1.w;
    a[8]  = (_Float16)f2.x; a[9]  = (_Float16)f2.y; a[10] = (_Float16)f2.z; a[11] = (_Float16)f2.w;
    a[12] = (_Float16)f3.x; a[13] = (_Float16)f3.y; a[14] = (_Float16)f3.z; a[15] = (_Float16)f3.w;
    return a;
}

// 16x32 f16 A-fragment straight from an f16 row.
__device__ __forceinline__ v16h load_a_f16(const _Float16* __restrict__ rowPtr, int klo) {
    const v8h lo = *(const v8h*)(rowPtr + klo);
    const v8h hh = *(const v8h*)(rowPtr + klo + 8);
    v16h a;
#pragma unroll
    for (int i = 0; i < 8; ++i) { a[i] = lo[i]; a[i + 8] = hh[i]; }
    return a;
}

// ---------------------------------------------------------------------------
__global__ void zero_ws_kernel(float4* __restrict__ p, long long n4) {
    long long i = (long long)blockIdx.x * blockDim.x + threadIdx.x;
    long long stride = (long long)gridDim.x * blockDim.x;
    float4 z; z.x = 0.f; z.y = 0.f; z.z = 0.f; z.w = 0.f;
    for (; i < n4; i += stride) p[i] = z;
}

// f16 B^T (N-major, contiguous-K) weight layouts.
//   W1t[h][i] = W1[i][h]            (128x1024)
//   W2n[h][o] = W2[h][o], o<1000    (128x1024, K-pad 0)  B^T for o@W2^T
//   W2t[o][h] = W2[h][o], o<1000    (1024x128, N-pad 0)  B^T for h@W2
__global__ void cvt_weights_kernel(const float* __restrict__ W1,
                                   const float* __restrict__ W2,
                                   _Float16* __restrict__ W1t,
                                   _Float16* __restrict__ W2n,
                                   _Float16* __restrict__ W2t) {
    int tid = blockIdx.x * blockDim.x + threadIdx.x;
    if (tid < H_SZ * I_SZ) {
        int h = tid >> 10, i = tid & 1023;
        W1t[tid] = (_Float16)W1[i * H_SZ + h];
    }
    if (tid < H_SZ * O_PAD) {
        int h = tid >> 10, o = tid & 1023;
        if (o < O_SZ) W2n[tid] = (_Float16)W2[h * O_SZ + o];
    }
    if (tid < O_PAD * H_SZ) {
        int o = tid >> 7, h = tid & 127;
        if (o < O_SZ) W2t[tid] = (_Float16)W2[h * O_SZ + o];
    }
}

// ---------------------------------------------------------------------------
// XW1 = x @ W1 (loop-invariant, one launch). block 128 = 4 waves.
__global__ void xw1_kernel(const float* __restrict__ x,
                           const _Float16* __restrict__ W1t,
                           float* __restrict__ XW1) {
    const int lane = threadIdx.x & 31, wave = threadIdx.x >> 5;
    const int l15 = lane & 15, hi = lane >> 4;
    const int mBase = (blockIdx.x * 4 + wave) * 16;

    v8f acc[8];
#pragma unroll
    for (int a = 0; a < 8; ++a) acc[a] = (v8f){0,0,0,0,0,0,0,0};

    const float* aRow = x + (long long)(mBase + l15) * I_SZ;
    for (int k0 = 0; k0 < I_SZ; k0 += 32) {
        const v16h a = load_a_from_f32(aRow, k0 + hi * 16);
#pragma unroll
        for (int nt = 0; nt < 8; ++nt) {
            const v16h b = *(const v16h*)(W1t + (nt * 16 + l15) * I_SZ + k0 + hi * 16);
            acc[nt] = __builtin_amdgcn_wmma_f32_16x16x32_f16(
                false, a, false, b, (short)0, acc[nt], false, false);
        }
    }
#pragma unroll
    for (int nt = 0; nt < 8; ++nt) {
        const int col = nt * 16 + l15;
#pragma unroll
        for (int r = 0; r < 8; ++r)
            XW1[(long long)(mBase + r + hi * 8) * H_SZ + col] = acc[nt][r];
    }
}

// ---------------------------------------------------------------------------
// h-update: T = o @ W2^T (K=O_PAD), gh = rho'(h)*(h-b_h-XW1-T),
//           hNew = clip(h - EPS*gh).  block 256 = 8 waves (M=128 rows/block).
// B K-slices (128 cols x 32 K) double-buffered in LDS via async copy.
#define HB_STRIDE 40   // 32 K-halfs + 8 pad (80B rows: spreads LDS banks)
__global__ void h_update_kernel(const _Float16* __restrict__ oF16, // [B][O_PAD]
                                const float* __restrict__ hOld,    // [B][H]
                                const float* __restrict__ XW1,     // [B][H]
                                const _Float16* __restrict__ W2n,  // [H][O_PAD]
                                const float* __restrict__ b_h,
                                float* __restrict__ hNew,
                                _Float16* __restrict__ hNewF16) {
    __shared__ _Float16 ldsB[2][H_SZ * HB_STRIDE];
    const int tid = threadIdx.x;
    const int lane = tid & 31, wave = tid >> 5;
    const int l15 = lane & 15, hi = lane >> 4;
    const int mBase = (blockIdx.x * 8 + wave) * 16;
    const unsigned ldsBase = (unsigned)(uintptr_t)(&ldsB[0][0]);
    const uint64_t w2nB = (uint64_t)(uintptr_t)W2n;

    // Stage one 128x32 slice of W2n: 8KB = 512 b128, 2 per thread.
    auto stage = [&](int buf, int k0) {
#pragma unroll
        for (int j = 0; j < 2; ++j) {
            const int u = tid * 2 + j;
            const int ncol = u >> 2, part = u & 3;
            const unsigned gOff = (unsigned)((ncol * O_PAD + k0 + part * 8) * 2);
            const unsigned lOff = ldsBase +
                (unsigned)(((buf * H_SZ + ncol) * HB_STRIDE + part * 8) * 2);
            async_b128(lOff, gOff, w2nB);
        }
    };

    v8f acc[8];
#pragma unroll
    for (int a = 0; a < 8; ++a) acc[a] = (v8f){0,0,0,0,0,0,0,0};

    stage(0, 0);
    wait_async0();
    __syncthreads();

    const _Float16* aRow = oF16 + (long long)(mBase + l15) * O_PAD;
    int buf = 0;
    for (int step = 0; step < O_PAD / 32; ++step) {
        const int k0 = step * 32;
        if (step + 1 < O_PAD / 32) stage(buf ^ 1, k0 + 32);

        const v16h a = load_a_f16(aRow, k0 + hi * 16);
        v16h bfrag[8];
#pragma unroll
        for (int nt = 0; nt < 8; ++nt)
            bfrag[nt] = *(const v16h*)&ldsB[buf][(nt * 16 + l15) * HB_STRIDE + hi * 16];
#pragma unroll
        for (int nt = 0; nt < 8; ++nt)
            acc[nt] = __builtin_amdgcn_wmma_f32_16x16x32_f16(
                false, a, false, bfrag[nt], (short)0, acc[nt], false, false);

        wait_async0();
        __syncthreads();
        buf ^= 1;
    }

#pragma unroll
    for (int nt = 0; nt < 8; ++nt) {
        const int col = nt * 16 + l15;
        const float bh = b_h[col];
#pragma unroll
        for (int r = 0; r < 8; ++r) {
            const long long idx = (long long)(mBase + r + hi * 8) * H_SZ + col;
            const float h  = hOld[idx];
            const float d  = h - bh - XW1[idx] - acc[nt][r];
            float hn = h - EPS_LR * (rho_prime(h) * d);
            hn = fminf(fmaxf(hn, 0.0f), 1.0f);
            hNew[idx] = hn;
            hNewF16[idx] = (_Float16)hn;
        }
    }
}

// ---------------------------------------------------------------------------
// o-update: U = h @ W2 (K=128), go = rho'(o)*(o-b_o-U), oNew = clip(o-EPS*go).
// block 256 = 8 waves; block tile = 128 rows x 64 cols; whole B tile in LDS.
#define OB_STRIDE 136  // 128 K-halfs + 8 pad
__global__ void o_update_kernel(const _Float16* __restrict__ hF16, // [B][H]
                                const float* __restrict__ oOld,    // [B][O_PAD]
                                const _Float16* __restrict__ W2t,  // [O_PAD][H]
                                const float* __restrict__ b_o,
                                float* __restrict__ oNew,
                                _Float16* __restrict__ oNewF16) {
    __shared__ _Float16 ldsB[64 * OB_STRIDE];
    const int tid = threadIdx.x;
    const int lane = tid & 31, wave = tid >> 5;
    const int l15 = lane & 15, hi = lane >> 4;
    const int mBase = (blockIdx.y * 8 + wave) * 16;
    const int nBase = blockIdx.x * 64;
    const unsigned ldsBase = (unsigned)(uintptr_t)(&ldsB[0]);
    const uint64_t w2tB = (uint64_t)(uintptr_t)W2t;

    // Stage 64 rows x 128 K of W2t: 16KB = 1024 b128, 4 per thread.
#pragma unroll
    for (int j = 0; j < 4; ++j) {
        const int u = tid * 4 + j;
        const int row = u >> 4, part = u & 15;
        const unsigned gOff = (unsigned)(((nBase + row) * H_SZ + part * 8) * 2);
        const unsigned lOff = ldsBase + (unsigned)((row * OB_STRIDE + part * 8) * 2);
        async_b128(lOff, gOff, w2tB);
    }

    v8f acc[4];
#pragma unroll
    for (int a = 0; a < 4; ++a) acc[a] = (v8f){0,0,0,0,0,0,0,0};

    wait_async0();
    __syncthreads();

    const _Float16* aRow = hF16 + (long long)(mBase + l15) * H_SZ;
#pragma unroll
    for (int k0 = 0; k0 < H_SZ; k0 += 32) {
        const v16h a = load_a_f16(aRow, k0 + hi * 16);
        v16h bfrag[4];
#pragma unroll
        for (int nt = 0; nt < 4; ++nt)
            bfrag[nt] = *(const v16h*)&ldsB[(nt * 16 + l15) * OB_STRIDE + k0 + hi * 16];
#pragma unroll
        for (int nt = 0; nt < 4; ++nt)
            acc[nt] = __builtin_amdgcn_wmma_f32_16x16x32_f16(
                false, a, false, bfrag[nt], (short)0, acc[nt], false, false);
    }

#pragma unroll
    for (int nt = 0; nt < 4; ++nt) {
        const int col = nBase + nt * 16 + l15;
        if (col < O_SZ) {
            const float bo = b_o[col];
#pragma unroll
            for (int r = 0; r < 8; ++r) {
                const long long idx = (long long)(mBase + r + hi * 8) * O_PAD + col;
                const float o = oOld[idx];
                const float d = o - bo - acc[nt][r];
                float on = o - EPS_LR * (rho_prime(o) * d);
                on = fminf(fmaxf(on, 0.0f), 1.0f);
                oNew[idx] = on;
                oNewF16[idx] = (_Float16)on;
            }
        }
    }
}

// ---------------------------------------------------------------------------
__global__ void log_softmax_kernel(const float* __restrict__ o, // [B][O_PAD]
                                   float* __restrict__ out) {   // [B][O]
    const int lane = threadIdx.x & 31;
    const int row  = blockIdx.x * (blockDim.x >> 5) + (threadIdx.x >> 5);
    const float* r = o + (long long)row * O_PAD;

    float m = -INFINITY;
    for (int c = lane; c < O_SZ; c += 32) m = fmaxf(m, r[c]);
#pragma unroll
    for (int off = 16; off > 0; off >>= 1) m = fmaxf(m, __shfl_xor(m, off, 32));

    float s = 0.0f;
    for (int c = lane; c < O_SZ; c += 32) s += expf(r[c] - m);
#pragma unroll
    for (int off = 16; off > 0; off >>= 1) s += __shfl_xor(s, off, 32);

    const float ls = m + logf(s);
    for (int c = lane; c < O_SZ; c += 32) out[(long long)row * O_SZ + c] = r[c] - ls;
}

// ---------------------------------------------------------------------------
extern "C" void kernel_launch(void* const* d_in, const int* in_sizes, int n_in,
                              void* d_out, int out_size, void* d_ws, size_t ws_size,
                              hipStream_t stream) {
    const float* x   = (const float*)d_in[0];
    // d_in[1]=hidden0 (zeros), d_in[2]=output0 (zeros), d_in[3]=b_in (not in grads)
    const float* b_h = (const float*)d_in[4];
    const float* b_o = (const float*)d_in[5];
    const float* W1  = (const float*)d_in[6];
    const float* W2  = (const float*)d_in[7];
    const int n_iterations = 30;  // d_in[8] is a device scalar; fixed by setup

    char* ws = (char*)d_ws;
    size_t off = 0;
    float* o0  = (float*)(ws + off); off += (size_t)B_SZ * O_PAD * sizeof(float);
    float* o1  = (float*)(ws + off); off += (size_t)B_SZ * O_PAD * sizeof(float);
    float* h0  = (float*)(ws + off); off += (size_t)B_SZ * H_SZ * sizeof(float);
    float* h1  = (float*)(ws + off); off += (size_t)B_SZ * H_SZ * sizeof(float);
    float* XW1 = (float*)(ws + off); off += (size_t)B_SZ * H_SZ * sizeof(float);
    _Float16* o0f = (_Float16*)(ws + off); off += (size_t)B_SZ * O_PAD * sizeof(_Float16);
    _Float16* o1f = (_Float16*)(ws + off); off += (size_t)B_SZ * O_PAD * sizeof(_Float16);
    _Float16* h0f = (_Float16*)(ws + off); off += (size_t)B_SZ * H_SZ * sizeof(_Float16);
    _Float16* h1f = (_Float16*)(ws + off); off += (size_t)B_SZ * H_SZ * sizeof(_Float16);
    _Float16* W1t = (_Float16*)(ws + off); off += (size_t)H_SZ * I_SZ * sizeof(_Float16);
    _Float16* W2n = (_Float16*)(ws + off); off += (size_t)H_SZ * O_PAD * sizeof(_Float16);
    _Float16* W2t = (_Float16*)(ws + off); off += (size_t)O_PAD * H_SZ * sizeof(_Float16);

    // Zero states (hidden0/output0 are zeros) + padded f16 mirrors/weights.
    zero_ws_kernel<<<2048, 256, 0, stream>>>((float4*)ws, (long long)(off / 16));

    cvt_weights_kernel<<<(H_SZ * O_PAD + 255) / 256, 256, 0, stream>>>(W1, W2, W1t, W2n, W2t);

    xw1_kernel<<<B_SZ / 64, 128, 0, stream>>>(x, W1t, XW1);

    float *hc = h0, *hn = h1, *oc = o0, *on = o1;
    _Float16 *hcf = h0f, *hnf = h1f, *ocf = o0f, *onf = o1f;
    for (int t = 0; t < n_iterations; ++t) {
        // Both kernels read OLD (hc,oc,hcf,ocf), write NEW buffers.
        h_update_kernel<<<B_SZ / 128, 256, 0, stream>>>(ocf, hc, XW1, W2n, b_h, hn, hnf);
        o_update_kernel<<<dim3(O_PAD / 64, B_SZ / 128), 256, 0, stream>>>(hcf, oc, W2t, b_o, on, onf);
        float* tf;  _Float16* tg;
        tf = hc; hc = hn; hn = tf;   tf = oc; oc = on; on = tf;
        tg = hcf; hcf = hnf; hnf = tg;   tg = ocf; ocf = onf; onf = tg;
    }

    log_softmax_kernel<<<B_SZ / 8, 256, 0, stream>>>(oc, (float*)d_out);
}
